// RNN_Model_31688268709894
// MI455X (gfx1250) — matrix operand order repaired and checked by reading, not compiled, not run on hardware
//
#include <hip/hip_runtime.h>
#include <hip/hip_bf16.h>
#include <math.h>

// Shapes from the reference
#define EMB 512
#define HID 1024
#define BB  64
#define SS  512
#define NBLK 32u   // persistent scan grid size (4 M-groups x 8 N-groups)

typedef __attribute__((ext_vector_type(16))) __bf16 v16bf;
typedef __attribute__((ext_vector_type(8)))  float  v8f;
typedef __attribute__((ext_vector_type(2)))  float  v2f;
typedef __attribute__((ext_vector_type(4)))  float  f4;

// ---------------------------------------------------------------------------
// Zero the t0 carry buffer + grid-barrier state
// ---------------------------------------------------------------------------
__global__ __launch_bounds__(256) void rnn_init_kernel(float* __restrict__ t0,
                                                       unsigned* __restrict__ bar) {
    int idx = blockIdx.x * 256 + threadIdx.x;
    t0[idx] = 0.0f;
    if (idx == 0) { bar[0] = 0u; bar[1] = 0u; }
}

// ---------------------------------------------------------------------------
// Phase 1: h_all = x @ W^T + bW   (M=32768, N=1024, K=512)
// bf16 WMMA, fp32 accumulate. Output written into the O region of d_out.
// ---------------------------------------------------------------------------
__global__ __launch_bounds__(256) void rnn_proj_kernel(const float* __restrict__ x,
                                                       const float* __restrict__ W,
                                                       const float* __restrict__ bW,
                                                       float* __restrict__ hall) {
    const int wid  = (blockIdx.x << 3) + (threadIdx.x >> 5); // global wave id
    const int lane = threadIdx.x & 31;
    const int tN   = wid & 63;   // 64 tiles over N=1024
    const int tM   = wid >> 6;   // 2048 tiles over M=32768
    const int m    = lane & 15;
    const int half = lane >> 4;

    const float* arow = x + (size_t)(tM * 16 + m) * EMB; // A row (x, row-major)
    const float* brow = W + (size_t)(tN * 16 + m) * EMB; // B col n = W row n

    v8f acc = {};
    for (int kc = 0; kc < EMB; kc += 32) {
        // A fragment (16x32 bf16): lane holds 16 consecutive K
        const f4* pa = (const f4*)(arow + kc + half * 16);
        f4 a0 = pa[0], a1 = pa[1], a2 = pa[2], a3 = pa[3];
        float af[16] = {a0.x, a0.y, a0.z, a0.w, a1.x, a1.y, a1.z, a1.w,
                        a2.x, a2.y, a2.z, a2.w, a3.x, a3.y, a3.z, a3.w};
        // B fragment (32x16 bf16): lane holds 16 consecutive K
        const f4* pb = (const f4*)(brow + kc + half * 16);
        f4 b0 = pb[0], b1 = pb[1], b2 = pb[2], b3 = pb[3];
        float bfv[16] = {b0.x, b0.y, b0.z, b0.w, b1.x, b1.y, b1.z, b1.w,
                         b2.x, b2.y, b2.z, b2.w, b3.x, b3.y, b3.z, b3.w};
        v16bf a, b;
#pragma unroll
        for (int e = 0; e < 16; ++e) {
            a[e] = (__bf16)af[e];
            b[e] = (__bf16)bfv[e];
        }
        acc = __builtin_amdgcn_wmma_f32_16x16x32_bf16(
            false, a, false, b, (short)0, acc, false, false);
    }

    const int   col  = tN * 16 + m;
    const float bias = bW[col];
#pragma unroll
    for (int r = 0; r < 8; ++r) {
        int row = tM * 16 + half * 8 + r;
        hall[(size_t)row * HID + col] = acc[r] + bias;
    }
}

// ---------------------------------------------------------------------------
// Software grid barrier (all NBLK blocks co-resident; sense via generation)
// ---------------------------------------------------------------------------
__device__ __forceinline__ void grid_barrier(unsigned* bar) {
    __syncthreads();
    if (threadIdx.x == 0) {
        __threadfence();
        unsigned gen = __hip_atomic_load(&bar[1], __ATOMIC_ACQUIRE,
                                         __HIP_MEMORY_SCOPE_AGENT);
        unsigned arrived = __hip_atomic_fetch_add(&bar[0], 1u, __ATOMIC_ACQ_REL,
                                                  __HIP_MEMORY_SCOPE_AGENT);
        if (arrived == NBLK - 1u) {
            __hip_atomic_store(&bar[0], 0u, __ATOMIC_RELAXED,
                               __HIP_MEMORY_SCOPE_AGENT);
            __hip_atomic_fetch_add(&bar[1], 1u, __ATOMIC_RELEASE,
                                   __HIP_MEMORY_SCOPE_AGENT);
        } else {
            while (__hip_atomic_load(&bar[1], __ATOMIC_ACQUIRE,
                                     __HIP_MEMORY_SCOPE_AGENT) == gen) {
                __builtin_amdgcn_s_sleep(2);
            }
        }
    }
    __syncthreads();
}

// ---------------------------------------------------------------------------
// Phase 2: persistent scan over all 512 timesteps. Per step:
//   h_    = tanh(0.6 * h_all[:, s, :] + 0.4 * t_prev)   (LDS, XOR-swizzled)
//   t_new = h_ @ U^T + bU                               (fp32 WMMA 16x16x4)
//   O[:, s-1, :] = t_prev
// 4 independent accumulators break the WMMA dependency chain.
// ---------------------------------------------------------------------------
__global__ __launch_bounds__(256) void rnn_scan_kernel(const float* __restrict__ U,
                                                       const float* __restrict__ bU,
                                                       float* __restrict__ t0,
                                                       float* __restrict__ t1,
                                                       float* __restrict__ O,
                                                       unsigned* __restrict__ bar) {
    __shared__ float hsm[16 * 1024]; // 64 KB
    const int mg  = blockIdx.x >> 3; // 0..3  (16 batch rows each)
    const int ng  = blockIdx.x & 7;  // 0..7  (128 cols each)
    const int tid = threadIdx.x;
    const int wv = tid >> 5, lane = tid & 31;
    const int m = lane & 15, half = lane >> 4;
    const int ncol = ng * 128 + wv * 16 + m;    // B/D column for this lane
    const float* urow = U + (size_t)ncol * HID; // U row n = B column
    const float  bias = bU[ncol];

    for (int s = 0; s < SS; ++s) {
        const float* tprev = (s & 1) ? t1 : t0;
        float*       tnext = (s & 1) ? t0 : t1;

        // commit previous carry into O slice s-1 (this block's 16x128 patch)
        if (s > 0) {
#pragma unroll
            for (int i = 0; i < 8; ++i) {
                int idx = tid + (i << 8);
                int row = idx >> 7, col = idx & 127;
                int b = mg * 16 + row, c = ng * 128 + col;
                O[((size_t)b * SS + (s - 1)) * HID + c] = tprev[b * HID + c];
            }
        }

        // h_ = tanh(0.6*h_all + 0.4*t_prev) into LDS (xor-swizzled per row)
        for (int i = 0; i < 64; ++i) {
            int idx = tid + (i << 8);
            int mm = idx >> 10, k = idx & 1023;
            int b  = mg * 16 + mm;
            float hv = 0.6f * O[((size_t)b * SS + s) * HID + k]
                     + 0.4f * tprev[b * HID + k];
            hsm[mm * 1024 + (k ^ (mm << 2))] = tanhf(hv);
        }
        __syncthreads();

        // warm L2->WGP path for next timestep's h_all rows (global_prefetch_b8)
        if (wv == 0 && s + 1 < SS) {
            __builtin_prefetch(
                O + ((size_t)(mg * 16 + m) * SS + (s + 1)) * HID + half * 512,
                0, 1);
        }

        // GEMM: one 16x16 tile per wave, 4 independent WMMA chains over K
        v8f acc0 = {}, acc1 = {}, acc2 = {}, acc3 = {};
        for (int k0 = 0; k0 < HID; k0 += 16) {
#pragma unroll
            for (int j = 0; j < 4; ++j) {
                int kb = k0 + j * 4 + half * 2; // A/B hold K = half*2 + r
                v2f a, b;
                a.x = hsm[m * 1024 + ((kb)     ^ (m << 2))];
                a.y = hsm[m * 1024 + ((kb + 1) ^ (m << 2))];
                b.x = urow[kb];
                b.y = urow[kb + 1];
                if (j == 0)      acc0 = __builtin_amdgcn_wmma_f32_16x16x4_f32(
                                     false, a, false, b, (short)0, acc0, false, false);
                else if (j == 1) acc1 = __builtin_amdgcn_wmma_f32_16x16x4_f32(
                                     false, a, false, b, (short)0, acc1, false, false);
                else if (j == 2) acc2 = __builtin_amdgcn_wmma_f32_16x16x4_f32(
                                     false, a, false, b, (short)0, acc2, false, false);
                else             acc3 = __builtin_amdgcn_wmma_f32_16x16x4_f32(
                                     false, a, false, b, (short)0, acc3, false, false);
            }
        }
        v8f acc = (acc0 + acc1) + (acc2 + acc3);

#pragma unroll
        for (int r = 0; r < 8; ++r) {
            int brow = mg * 16 + half * 8 + r;
            tnext[brow * HID + ncol] = acc[r] + bias;
        }

        // all blocks must finish writing tnext (and O slice) before next step
        grid_barrier(bar);
    }
}

// ---------------------------------------------------------------------------
// Tail: O[:, S-1, :] = t_final and d_out[0:B*H] = t_final
// ---------------------------------------------------------------------------
__global__ __launch_bounds__(256) void rnn_tail_kernel(const float* __restrict__ tfin,
                                                       float* __restrict__ out_t,
                                                       float* __restrict__ O) {
    int idx = blockIdx.x * 256 + threadIdx.x; // 0..65535
    float v = tfin[idx];
    out_t[idx] = v;
    int b = idx >> 10, c = idx & 1023;
    O[((size_t)b * SS + (SS - 1)) * HID + c] = v;
}

// ---------------------------------------------------------------------------
extern "C" void kernel_launch(void* const* d_in, const int* in_sizes, int n_in,
                              void* d_out, int out_size, void* d_ws, size_t ws_size,
                              hipStream_t stream) {
    const float* x  = (const float*)d_in[0]; // [B,S,EMB]
    const float* W  = (const float*)d_in[1]; // [HID,EMB]
    const float* bW = (const float*)d_in[2]; // [HID]
    const float* U  = (const float*)d_in[3]; // [HID,HID]
    const float* bU = (const float*)d_in[4]; // [HID]

    float* out_t = (float*)d_out;                    // t_final [B,HID]
    float* O     = (float*)d_out + (size_t)BB * HID; // O [B,S,HID]

    float*    t0  = (float*)d_ws;            // 64*1024 floats
    float*    t1  = t0 + BB * HID;           // 64*1024 floats
    unsigned* bar = (unsigned*)(t1 + BB * HID);

    rnn_init_kernel<<<256, 256, 0, stream>>>(t0, bar);

    // h_all written straight into O region (overwritten slice-by-slice later)
    rnn_proj_kernel<<<16384, 256, 0, stream>>>(x, W, bW, O);

    // single persistent launch replaces 1024 per-step launches
    rnn_scan_kernel<<<NBLK, 256, 0, stream>>>(U, bU, t0, t1, O, bar);

    // 512 steps -> final carry lives in t0
    rnn_tail_kernel<<<256, 256, 0, stream>>>(t0, out_t, O);
}
